// model_memory_single_10909216932232
// MI455X (gfx1250) — matrix operand order repaired and checked by reading, not compiled, not run on hardware
//
#include <hip/hip_runtime.h>
#include <hip/hip_bf16.h>

#define DIM 48
#define B_SZ 128
#define PLEN 20
#define FLEN 40
#define NPRED 5
#define NMEM 200000

typedef _Float16 v16h __attribute__((ext_vector_type(16)));
typedef float    v8f  __attribute__((ext_vector_type(8)));

// ---- workspace byte offsets ----
#define OFF_STATE 0            // 128*48 f32            (24576 B)
#define OFF_SNORM 24576        // 128*64 f16            (16384 B)
#define OFF_GBEST 40960        // 128 u64               (1024 B)
#define OFF_SEL   41984        // 128*5 int             (2560 B)
#define OFF_INFO  45056        // 5*128*96 f32          (245760 B)
#define OFF_PRED  290816       // 5*128*40*2 f32        (204800 B)
#define OFF_FEAT  495616       // 5*128*40*32 f32       (3276800 B)

__device__ __forceinline__ float sigmoidf_(float x) { return 1.0f / (1.0f + expf(-x)); }

// ======================= 1. past conv + encoder GRU + l2norm =======================
__global__ void enc_kernel(const float* __restrict__ past, const float* __restrict__ cw,
                           const float* __restrict__ cb, const float* __restrict__ wih,
                           const float* __restrict__ whh, const float* __restrict__ bih,
                           const float* __restrict__ bhh, float* __restrict__ stateP,
                           _Float16* __restrict__ snorm, unsigned long long* __restrict__ gbest) {
  int b = blockIdx.x, tid = threadIdx.x;
  __shared__ float xc[PLEN][DIM];
  __shared__ float gi[3 * DIM], gh[3 * DIM], h[DIM];
  __shared__ float nrm;
  for (int u = tid; u < PLEN * DIM; u += blockDim.x) {
    int t = u / DIM, c = u % DIM;
    float acc = cb[c];
    for (int ic = 0; ic < 2; ++ic)
      for (int kk = 0; kk < 3; ++kk) {
        int tt = t + kk - 1;
        if (tt >= 0 && tt < PLEN) acc += past[(b * PLEN + tt) * 2 + ic] * cw[(c * 2 + ic) * 3 + kk];
      }
    xc[t][c] = fmaxf(acc, 0.0f);
  }
  if (tid < DIM) h[tid] = 0.0f;
  __syncthreads();
  for (int t = 0; t < PLEN; ++t) {
    if (tid < 3 * DIM) {
      float a = bih[tid], g = bhh[tid];
      const float* wi = wih + tid * DIM;
      const float* wh = whh + tid * DIM;
      for (int q = 0; q < DIM; ++q) { a += xc[t][q] * wi[q]; g += h[q] * wh[q]; }
      gi[tid] = a; gh[tid] = g;
    }
    __syncthreads();
    if (tid < DIM) {
      float r = sigmoidf_(gi[tid] + gh[tid]);
      float z = sigmoidf_(gi[DIM + tid] + gh[DIM + tid]);
      float n = tanhf(gi[2 * DIM + tid] + r * gh[2 * DIM + tid]);
      h[tid] = (1.0f - z) * n + z * h[tid];
    }
    __syncthreads();
  }
  if (tid == 0) {
    float s = 0.0f;
    for (int j = 0; j < DIM; ++j) s += h[j] * h[j];
    nrm = fmaxf(sqrtf(s), 1e-12f);
    gbest[b] = 0ull;
  }
  if (tid < DIM) stateP[b * DIM + tid] = h[tid];
  __syncthreads();
  if (tid < 64) snorm[b * 64 + tid] = (tid < DIM) ? (_Float16)(h[tid] / nrm) : (_Float16)0.0f;
}

// ======================= 2. cosine scores via WMMA + top-1 (per pass) =======================
// Staging uses GLOBAL_LOAD_ASYNC_TO_LDS_B128 (ASYNCcnt path): memory -> LDS without VGPRs,
// consecutive lanes copy consecutive 16B chunks (fully coalesced), single s_wait_asynccnt.
__global__ void score_kernel(const float* __restrict__ mem, const _Float16* __restrict__ snorm,
                             const int* __restrict__ sel, unsigned long long* __restrict__ gbest,
                             int pass) {
  __shared__ __align__(16) _Float16 A[B_SZ][64];   // normalized states, K padded to 64 (16 KB)
  __shared__ __align__(16) _Float16 R[64][64];     // normalized memory rows f16 (8 KB)
  __shared__ __align__(16) float R32[64 * DIM];    // raw memory rows f32 (12 KB)
  __shared__ unsigned long long best[B_SZ];
  int tid = threadIdx.x;
  int row0 = blockIdx.x * 64;

  // async copy: state matrix (1024 x 16B) and this block's 64 memory rows (768 x 16B)
  {
    unsigned la = (unsigned)(size_t)(&A[0][0]);
    const char* ga = (const char*)snorm;
    for (int u = tid; u < 1024; u += 256) {
      unsigned loff = la + u * 16;
      const char* g = ga + (size_t)u * 16;
      asm volatile("global_load_async_to_lds_b128 %0, %1, off" :: "v"(loff), "v"(g) : "memory");
    }
    unsigned lrr = (unsigned)(size_t)(&R32[0]);
    const char* grr = (const char*)(mem + (size_t)row0 * DIM);
    for (int u = tid; u < 768; u += 256) {
      unsigned loff = lrr + u * 16;
      const char* g = grr + (size_t)u * 16;
      asm volatile("global_load_async_to_lds_b128 %0, %1, off" :: "v"(loff), "v"(g) : "memory");
    }
  }
  if (tid < B_SZ) best[tid] = 0ull;
  {
    int pr = row0 + 64 + tid;
    if (tid < 64 && pr < NMEM) __builtin_prefetch(mem + (size_t)pr * DIM, 0, 0);
  }
  asm volatile("s_wait_asynccnt 0x0" ::: "memory");
  __syncthreads();

  if (tid < 64) {
    const float* v = &R32[tid * DIM];
    float s = 0.0f;
    for (int j = 0; j < DIM; ++j) s += v[j] * v[j];
    float inv = 1.0f / fmaxf(sqrtf(s), 1e-12f);
    for (int j = 0; j < DIM; ++j) R[tid][j] = (_Float16)(v[j] * inv);
    for (int j = DIM; j < 64; ++j) R[tid][j] = (_Float16)0.0f;
  }
  __syncthreads();

  int wid = tid >> 5, lane = tid & 31, lr = lane & 15, hi = lane >> 4;
  // A fragments for both k-steps (16-bit A 16x32 layout)
  v16h a0, a1;
  int arow = wid * 16 + lr, off = hi * 8;
  for (int i = 0; i < 8; ++i) {
    a0[i] = A[arow][off + i];       a0[8 + i] = A[arow][16 + off + i];
    a1[i] = A[arow][32 + off + i];  a1[8 + i] = A[arow][48 + off + i];
  }
  for (int n = 0; n < 4; ++n) {
    int rrow = n * 16 + lr;
    v16h b0, b1;
    for (int i = 0; i < 16; ++i) { b0[i] = R[rrow][(i & 7) + 8 * hi + 16 * (i >> 3)]; b1[i] = R[rrow][32 + (i & 7) + 8 * hi + 16 * (i >> 3)]; }
    v8f c = {};
    c = __builtin_amdgcn_wmma_f32_16x16x32_f16(false, a0, false, b0, (short)0, c, false, false);
    c = __builtin_amdgcn_wmma_f32_16x16x32_f16(false, a1, false, b1, (short)0, c, false, false);
    int row = row0 + n * 16 + lr;
    for (int r8 = 0; r8 < 8; ++r8) {
      int batch = wid * 16 + r8 + 8 * hi;
      float s = c[r8];
      bool skip = false;
      for (int j = 0; j < pass; ++j)
        if (sel[batch * NPRED + j] == row) skip = true;
      if (!skip) {
        unsigned u = __float_as_uint(s);
        u = (u & 0x80000000u) ? ~u : (u | 0x80000000u);
        unsigned long long pk = ((unsigned long long)u << 32) | (unsigned)row;
        atomicMax(&best[batch], pk);
      }
    }
  }
  __syncthreads();
  if (tid < B_SZ) atomicMax(&gbest[tid], best[tid]);
}

// ======================= 3. select winner, build decoder input =======================
__global__ void select_kernel(const float* __restrict__ mfut, const float* __restrict__ stateP,
                              unsigned long long* __restrict__ gbest, int* __restrict__ sel,
                              float* __restrict__ info, int pass) {
  int b = threadIdx.x;
  if (b >= B_SZ) return;
  unsigned long long pk = gbest[b];
  int idx = (int)(pk & 0xFFFFFFFFull);
  sel[b * NPRED + pass] = idx;
  gbest[b] = 0ull;
  float* ip = info + ((size_t)pass * B_SZ + b) * 96;
  const float* mf = mfut + (size_t)idx * DIM;
  for (int j = 0; j < DIM; ++j) { ip[j] = stateP[b * DIM + j]; ip[DIM + j] = mf[j]; }
}

// ======================= 4. decoder GRU (WMMA recurrent matmul) =======================
__global__ void dec_kernel(const float* __restrict__ info, const float* __restrict__ past,
                           const float* __restrict__ wih, const float* __restrict__ whh,
                           const float* __restrict__ bih, const float* __restrict__ bhh,
                           const float* __restrict__ fow, const float* __restrict__ fob,
                           float* __restrict__ pred) {
  const int HID = 96;
  int bg = blockIdx.x, k = blockIdx.y, b0 = bg * 32, tid = threadIdx.x;
  __shared__ float gh[32][288];
  __shared__ float hc[32][96];
  __shared__ _Float16 hf[32][96];
  __shared__ float pres[32][2];
  // t = 0: h0 == 0  ->  gh == b_hh; gi from info (done scalar)
  for (int u = tid; u < 32 * HID; u += 256) {
    int b = u / HID, j = u % HID;
    const float* ip = info + ((size_t)k * B_SZ + b0 + b) * 96;
    float gr = bih[j], gz = bih[HID + j], gn = bih[2 * HID + j];
    const float* wr = wih + (size_t)j * 96;
    const float* wz = wih + (size_t)(HID + j) * 96;
    const float* wn = wih + (size_t)(2 * HID + j) * 96;
    for (int q = 0; q < 96; ++q) { float x = ip[q]; gr += x * wr[q]; gz += x * wz[q]; gn += x * wn[q]; }
    float r = sigmoidf_(gr + bhh[j]);
    float z = sigmoidf_(gz + bhh[HID + j]);
    float n = tanhf(gn + r * bhh[2 * HID + j]);
    float h = (1.0f - z) * n;
    hc[b][j] = h; hf[b][j] = (_Float16)h;
  }
  for (int u = tid; u < 64; u += 256) {
    int b = u >> 1, c = u & 1;
    pres[b][c] = past[((b0 + b) * PLEN + PLEN - 1) * 2 + c];
  }
  __syncthreads();
  for (int u = tid; u < 64; u += 256) {
    int b = u >> 1, c = u & 1;
    float acc = pres[b][c] + fob[c];
    for (int j = 0; j < HID; ++j) acc += hc[b][j] * fow[c * HID + j];
    pred[(((size_t)k * B_SZ + b0 + b) * FLEN + 0) * 2 + c] = acc;
    pres[b][c] = acc;
  }
  __syncthreads();
  int wid = tid >> 5, lane = tid & 31, lr = lane & 15, hi = lane >> 4;
  for (int t = 1; t < FLEN; ++t) {
    // gh = h @ whh^T  (M=32 batches, N=288 gates, K=96) via WMMA f16->f32
    for (int tile = wid; tile < 36; tile += 8) {
      int m = tile / 18, n = tile % 18;
      v8f c = {};
      for (int ks = 0; ks < 3; ++ks) {
        int kb = ks * 32;
        v16h a, bf;
        int ar = m * 16 + lr, offA = hi * 8;
        for (int i = 0; i < 8; ++i) {
          a[i]     = hf[ar][kb + offA + i];
          a[8 + i] = hf[ar][kb + 16 + offA + i];
        }
        int gRow = n * 16 + lr;
        const float* wp = whh + (size_t)gRow * 96 + kb + 8 * hi;
        for (int i = 0; i < 16; ++i) bf[i] = (_Float16)wp[(i & 7) + 16 * (i >> 3)];
        c = __builtin_amdgcn_wmma_f32_16x16x32_f16(false, a, false, bf, (short)0, c, false, false);
      }
      for (int r8 = 0; r8 < 8; ++r8) gh[m * 16 + r8 + 8 * hi][n * 16 + lr] = c[r8];
    }
    __syncthreads();
    for (int u = tid; u < 32 * HID; u += 256) {
      int b = u / HID, j = u % HID;
      float r = sigmoidf_(bih[j] + gh[b][j] + bhh[j]);
      float z = sigmoidf_(bih[HID + j] + gh[b][HID + j] + bhh[HID + j]);
      float n = tanhf(bih[2 * HID + j] + r * (gh[b][2 * HID + j] + bhh[2 * HID + j]));
      float h = (1.0f - z) * n + z * hc[b][j];
      hc[b][j] = h; hf[b][j] = (_Float16)h;
    }
    __syncthreads();
    for (int u = tid; u < 64; u += 256) {
      int b = u >> 1, c2 = u & 1;
      float acc = pres[b][c2] + fob[c2];
      for (int j = 0; j < HID; ++j) acc += hc[b][j] * fow[c2 * HID + j];
      pred[(((size_t)k * B_SZ + b0 + b) * FLEN + t) * 2 + c2] = acc;
      pres[b][c2] = acc;
    }
    __syncthreads();
  }
}

// ======================= 5. on-the-fly conv1+conv2 at sampled pixels =======================
__global__ void feat_kernel(const float* __restrict__ scene, const float* __restrict__ c1w,
                            const float* __restrict__ c1b, const float* __restrict__ c2w,
                            const float* __restrict__ c2b, const float* __restrict__ pred,
                            float* __restrict__ feat) {
  int b = blockIdx.x, k = blockIdx.y, tid = threadIdx.x;
  __shared__ float p1[25][16];
  __shared__ int s_cx, s_cy, s_valid;
  for (int t = 0; t < FLEN; ++t) {
    if (tid == 0) {
      float px = pred[(((size_t)k * B_SZ + b) * FLEN + t) * 2 + 0];
      float py = pred[(((size_t)k * B_SZ + b) * FLEN + t) * 2 + 1];
      float fx = rintf((px + 89.0f) * 0.5f);
      float fy = rintf((py + 89.0f) * 0.5f);
      s_valid = (fx >= 0.0f && fx < 90.0f && fy >= 0.0f && fy < 90.0f) ? 1 : 0;
      s_cx = (int)fminf(fmaxf(fx, 0.0f), 89.0f);
      s_cy = (int)fminf(fmaxf(fy, 0.0f), 89.0f);
    }
    __syncthreads();
    int valid = s_valid, cx = s_cx, cy = s_cy;
    for (int u = tid; u < 400; u += 256) {
      int pos = u / 16, ic = u % 16;
      int y = cy + pos / 5 - 2, x = cx + pos % 5 - 2;
      float val = 0.0f;
      if (valid && y >= 0 && y < 90 && x >= 0 && x < 90) {
        float acc = c1b[ic];
        for (int ky = 0; ky < 5; ++ky) {
          int sy = 2 * y + ky - 2;
          if (sy < 0 || sy >= 180) continue;
          for (int kx = 0; kx < 5; ++kx) {
            int sx = 2 * x + kx - 2;
            if (sx < 0 || sx >= 180) continue;
            const float* sp = scene + (((size_t)b * 180 + sy) * 180 + sx) * 4;
            for (int c = 0; c < 4; ++c) acc += sp[c] * c1w[((ic * 4 + c) * 5 + ky) * 5 + kx];
          }
        }
        val = fmaxf(acc, 0.0f);
      }
      p1[pos][ic] = val;
    }
    __syncthreads();
    if (tid < 32) {
      float f = 0.0f;
      if (valid) {
        float acc = c2b[tid];
        for (int ic = 0; ic < 16; ++ic)
          for (int pos = 0; pos < 25; ++pos)
            acc += p1[pos][ic] * c2w[((tid * 16 + ic) * 5 + pos / 5) * 5 + pos % 5];
        f = fmaxf(acc, 0.0f);
      }
      feat[((((size_t)k * B_SZ + b) * FLEN + t) * 32) + tid] = f;
    }
    __syncthreads();
  }
}

// ======================= 6. refinement GRU + pred update =======================
__global__ void refine_kernel(const float* __restrict__ feat, const float* __restrict__ stateP,
                              const float* __restrict__ wih, const float* __restrict__ whh,
                              const float* __restrict__ bih, const float* __restrict__ bhh,
                              const float* __restrict__ frw, const float* __restrict__ frb,
                              float* __restrict__ pred) {
  int bg = blockIdx.x, k = blockIdx.y, tid = threadIdx.x, b0 = bg * 16;
  __shared__ float h[16][DIM];
  __shared__ float ft[16][32];
  for (int u = tid; u < 16 * DIM; u += 256) h[u / DIM][u % DIM] = stateP[(b0 + u / DIM) * DIM + u % DIM];
  __syncthreads();
  for (int t = 0; t < FLEN; ++t) {
    for (int u = tid; u < 16 * 32; u += 256) {
      int b = u >> 5, q = u & 31;
      ft[b][q] = feat[((((size_t)k * B_SZ + b0 + b) * FLEN + t) * 32) + q];
    }
    __syncthreads();
    float hn[3];
    for (int s = 0; s < 3; ++s) {
      int u = tid + s * 256;
      int b = u / DIM, j = u % DIM;
      float gr = bih[j], gz = bih[DIM + j], gn = bih[2 * DIM + j];
      const float* wr = wih + j * 32;
      const float* wz = wih + (DIM + j) * 32;
      const float* wn = wih + (2 * DIM + j) * 32;
      for (int q = 0; q < 32; ++q) { float x = ft[b][q]; gr += x * wr[q]; gz += x * wz[q]; gn += x * wn[q]; }
      float hr = bhh[j], hz = bhh[DIM + j], hv = bhh[2 * DIM + j];
      const float* vr = whh + j * DIM;
      const float* vz = whh + (DIM + j) * DIM;
      const float* vn = whh + (2 * DIM + j) * DIM;
      for (int q = 0; q < DIM; ++q) { float x = h[b][q]; hr += x * vr[q]; hz += x * vz[q]; hv += x * vn[q]; }
      float r = sigmoidf_(gr + hr);
      float z = sigmoidf_(gz + hz);
      float n = tanhf(gn + r * hv);
      hn[s] = (1.0f - z) * n + z * h[b][j];
    }
    __syncthreads();
    for (int s = 0; s < 3; ++s) { int u = tid + s * 256; h[u / DIM][u % DIM] = hn[s]; }
    __syncthreads();
  }
  for (int u = tid; u < 16 * 80; u += 256) {
    int b = u / 80, o = u % 80;
    float acc = frb[o];
    for (int j = 0; j < DIM; ++j) acc += h[b][j] * frw[o * DIM + j];
    int t = o >> 1, c = o & 1;
    pred[(((size_t)k * B_SZ + b0 + b) * FLEN + t) * 2 + c] += acc;
  }
}

// ======================= 7. transpose preds into output layout =======================
__global__ void out_kernel(const float* __restrict__ pred, float* __restrict__ out) {
  int i = blockIdx.x * blockDim.x + threadIdx.x;
  if (i >= B_SZ * NPRED * FLEN * 2) return;
  int c = i & 1;
  int t = (i >> 1) % FLEN;
  int k = (i / (2 * FLEN)) % NPRED;
  int b = i / (2 * FLEN * NPRED);
  out[i] = pred[(((size_t)k * B_SZ + b) * FLEN + t) * 2 + c];
}

extern "C" void kernel_launch(void* const* d_in, const int* in_sizes, int n_in,
                              void* d_out, int out_size, void* d_ws, size_t ws_size,
                              hipStream_t stream) {
  (void)in_sizes; (void)n_in; (void)out_size; (void)ws_size;
  const float* past        = (const float*)d_in[0];
  const float* scene       = (const float*)d_in[1];
  const float* memory_past = (const float*)d_in[2];
  const float* memory_fut  = (const float*)d_in[3];
  const float* conv_past_w = (const float*)d_in[4];
  const float* conv_past_b = (const float*)d_in[5];
  const float* enc_w_ih    = (const float*)d_in[6];
  const float* enc_w_hh    = (const float*)d_in[7];
  const float* enc_b_ih    = (const float*)d_in[8];
  const float* enc_b_hh    = (const float*)d_in[9];
  const float* dec_w_ih    = (const float*)d_in[10];
  const float* dec_w_hh    = (const float*)d_in[11];
  const float* dec_b_ih    = (const float*)d_in[12];
  const float* dec_b_hh    = (const float*)d_in[13];
  const float* fc_out_w    = (const float*)d_in[14];
  const float* fc_out_b    = (const float*)d_in[15];
  const float* cs1_w       = (const float*)d_in[16];
  const float* cs1_b       = (const float*)d_in[17];
  const float* cs2_w       = (const float*)d_in[18];
  const float* cs2_b       = (const float*)d_in[19];
  const float* rnn_w_ih    = (const float*)d_in[20];
  const float* rnn_w_hh    = (const float*)d_in[21];
  const float* rnn_b_ih    = (const float*)d_in[22];
  const float* rnn_b_hh    = (const float*)d_in[23];
  const float* fc_ref_w    = (const float*)d_in[24];
  const float* fc_ref_b    = (const float*)d_in[25];

  char* ws = (char*)d_ws;
  float* stateP             = (float*)(ws + OFF_STATE);
  _Float16* snorm           = (_Float16*)(ws + OFF_SNORM);
  unsigned long long* gbest = (unsigned long long*)(ws + OFF_GBEST);
  int* sel                  = (int*)(ws + OFF_SEL);
  float* info               = (float*)(ws + OFF_INFO);
  float* pred               = (float*)(ws + OFF_PRED);
  float* feat               = (float*)(ws + OFF_FEAT);
  float* out                = (float*)d_out;

  enc_kernel<<<B_SZ, 160, 0, stream>>>(past, conv_past_w, conv_past_b, enc_w_ih, enc_w_hh,
                                       enc_b_ih, enc_b_hh, stateP, snorm, gbest);
  for (int pass = 0; pass < NPRED; ++pass) {
    score_kernel<<<NMEM / 64, 256, 0, stream>>>(memory_past, snorm, sel, gbest, pass);
    select_kernel<<<1, 128, 0, stream>>>(memory_fut, stateP, gbest, sel, info, pass);
  }
  dec_kernel<<<dim3(4, NPRED), 256, 0, stream>>>(info, past, dec_w_ih, dec_w_hh, dec_b_ih,
                                                 dec_b_hh, fc_out_w, fc_out_b, pred);
  for (int it = 0; it < 4; ++it) {
    feat_kernel<<<dim3(B_SZ, NPRED), 256, 0, stream>>>(scene, cs1_w, cs1_b, cs2_w, cs2_b, pred, feat);
    refine_kernel<<<dim3(8, NPRED), 256, 0, stream>>>(feat, stateP, rnn_w_ih, rnn_w_hh,
                                                      rnn_b_ih, rnn_b_hh, fc_ref_w, fc_ref_b, pred);
  }
  out_kernel<<<(B_SZ * NPRED * FLEN * 2 + 255) / 256, 256, 0, stream>>>(pred, out);
}
